// LSTM_91319594647936
// MI455X (gfx1250) — compile-verified, matrix operand order checked by reading, not run on hardware
//
#include <hip/hip_runtime.h>
#include <hip/hip_bf16.h>
#include <math.h>

// ---------------------------------------------------------------------------
// LSTM on MI455X (gfx1250): bf16 WMMA 16x16x32, f32 accumulate.
//   B=64, S=512, D=512, H=1024, gates = [x_t | h] @ [W_ih ; W_hh] + bias
// Weights pre-packed into the exact WMMA B-fragment VGPR layout so each wave
// reads contiguous 1KB fragments (two b128 loads per lane). Weights (12MB bf16)
// stay resident in the 192MB L2 across all 512 sequential steps.
// ---------------------------------------------------------------------------

typedef __attribute__((ext_vector_type(16))) __bf16 v16bf;
typedef __attribute__((ext_vector_type(8)))  __bf16 v8bf;
typedef __attribute__((ext_vector_type(8)))  float  v8f;

#define LB  64    // batch
#define LS  512   // sequence length
#define LD  512   // input dim
#define LH  1024  // hidden dim
#define LG  4096  // 4*H

// ---------------------------------------------------------------- prep kernels

__global__ void f32_to_bf16_kernel(const float* __restrict__ in,
                                   __bf16* __restrict__ out, size_t n) {
  size_t i = (size_t)blockIdx.x * blockDim.x + threadIdx.x;
  if (i < n) out[i] = (__bf16)in[i];
}

__global__ void zero_bf16_kernel(__bf16* __restrict__ out, size_t n) {
  size_t i = (size_t)blockIdx.x * blockDim.x + threadIdx.x;
  if (i < n) out[i] = (__bf16)0.0f;
}

__global__ void zero_f32_kernel(float* __restrict__ out, size_t n) {
  size_t i = (size_t)blockIdx.x * blockDim.x + threadIdx.x;
  if (i < n) out[i] = 0.0f;
}

// Pack W (K x 4096, row-major f32) into per-tile WMMA B fragments (bf16).
// Fragment (nt, kb) = 512 bf16 = 1KB, laid out as:
//   [half 0: lane 0..31 x 8 elems][half 1: lane 0..31 x 8 elems]
// lane l (<16): N = nt*16+l,    K = kb*32 + half*16 + 0..7
// lane l (>=16): N = nt*16+l-16, K = kb*32 + half*16 + 8..15
// matching the CDNA5 16-bit B-matrix 32x16 VGPR layout (ISA 7.12.2).
__global__ void pack_weights_kernel(const float* __restrict__ W,
                                    __bf16* __restrict__ out,
                                    int nkb, size_t total) {
  size_t p = (size_t)blockIdx.x * blockDim.x + threadIdx.x;
  if (p >= total) return;
  int    j    = (int)(p & 7);
  int    lane = (int)((p >> 3) & 31);
  int    half = (int)((p >> 8) & 1);
  size_t frag = p >> 9;
  int    kb   = (int)(frag % (size_t)nkb);
  int    nt   = (int)(frag / (size_t)nkb);
  int n = nt * 16 + (lane & 15);
  int k = kb * 32 + ((lane < 16) ? 0 : 8) + half * 16 + j;
  out[p] = (__bf16)W[(size_t)k * LG + n];
}

// ---------------------------------------------------------------- step kernel

__device__ __forceinline__ v16bf load_a_frag(const __bf16* __restrict__ row,
                                             int k0) {
  // CDNA5 16-bit A-matrix 16x32 layout: lane<16 reads K [k0..k0+7],[k0+16..k0+23]
  // lane>=16 reads K [k0+8..k0+15],[k0+24..k0+31]; k0 already lane-adjusted.
  v8bf lo = *(const v8bf*)(row + k0);
  v8bf hi = *(const v8bf*)(row + k0 + 16);
  v16bf a;
#pragma unroll
  for (int i = 0; i < 8; ++i) { a[i] = lo[i]; a[i + 8] = hi[i]; }
  return a;
}

__device__ __forceinline__ v16bf load_b_frag(const __bf16* __restrict__ frag,
                                             int lane) {
  v8bf lo = *(const v8bf*)(frag + lane * 8);
  v8bf hi = *(const v8bf*)(frag + 256 + lane * 8);
  v16bf b;
#pragma unroll
  for (int i = 0; i < 8; ++i) { b[i] = lo[i]; b[i + 8] = hi[i]; }
  return b;
}

__device__ __forceinline__ float sigmoidf_fast(float x) {
  return 1.0f / (1.0f + __expf(-x));
}

// One block = one (m_tile, h_tile) pair; 4 waves = gates i,f,g,o.
// Each wave: 48 x v_wmma_f32_16x16x32_bf16 over K = D + H = 1536.
__global__ __launch_bounds__(128)
void lstm_step_kernel(const __bf16* __restrict__ xb,     // (B,S,D) bf16
                      const __bf16* __restrict__ wih,    // packed (256 tiles x 16 kb)
                      const __bf16* __restrict__ whh,    // packed (256 tiles x 32 kb)
                      const float*  __restrict__ bias,   // (4096)
                      const __bf16* __restrict__ h_in,   // (B,H) bf16
                      __bf16*       __restrict__ h_out,  // (B,H) bf16
                      float*        __restrict__ c,      // (B,H) f32, in-place
                      float*        __restrict__ hseq,   // (B,S,H) f32
                      float*        __restrict__ h_final,// (B,H) f32 or null
                      int t) {
  const int tid  = threadIdx.x;
  const int gate = tid >> 5;          // wave id = gate id (i,f,g,o)
  const int lane = tid & 31;
  const int mj   = blockIdx.x;        // 0..255
  const int mt   = mj >> 6;           // batch tile 0..3
  const int j    = mj & 63;           // hidden tile 0..63
  const int m0   = mt * 16;
  const int nt   = gate * (LH / 16) + j;   // global column tile in (4096)

  const int arow = m0 + (lane & 15);       // batch row this lane serves
  const int ksel = (lane < 16) ? 0 : 8;

  v8f acc = {};

  // ---- x_t @ W_ih : K = 512 -> 16 k-blocks
  {
    const __bf16* xrow = xb + ((size_t)arow * LS + t) * LD;
    const __bf16* wf   = wih + ((size_t)nt * (LD / 32)) * 512;
#pragma unroll 4
    for (int kb = 0; kb < LD / 32; ++kb) {
      v16bf a = load_a_frag(xrow, kb * 32 + ksel);
      v16bf b = load_b_frag(wf + (size_t)kb * 512, lane);
      acc = __builtin_amdgcn_wmma_f32_16x16x32_bf16(
          false, a, false, b, (short)0, acc, false, false);
    }
  }

  // ---- h @ W_hh : K = 1024 -> 32 k-blocks (weights L2-resident)
  {
    const __bf16* hrow = h_in + (size_t)arow * LH;
    const __bf16* wf   = whh + ((size_t)nt * (LH / 32)) * 512;
#pragma unroll 4
    for (int kb = 0; kb < LH / 32; ++kb) {
      v16bf a = load_a_frag(hrow, kb * 32 + ksel);
      v16bf b = load_b_frag(wf + (size_t)kb * 512, lane);
      acc = __builtin_amdgcn_wmma_f32_16x16x32_bf16(
          false, a, false, b, (short)0, acc, false, false);
    }
  }

  // ---- bias: column = gate*H + j*16 + (lane&15), same for all 8 acc rows
  {
    float bv = bias[gate * LH + j * 16 + (lane & 15)];
#pragma unroll
    for (int v = 0; v < 8; ++v) acc[v] += bv;
  }

  // ---- exchange the 4 gate tiles through LDS, then do the cell update.
  // C/D layout: VGPR v, lane l -> M = v + 8*(l>=16), N = l&15.
  __shared__ float lds[4][8][32];
#pragma unroll
  for (int v = 0; v < 8; ++v) lds[gate][v][lane] = acc[v];
  __syncthreads();

  for (int e = tid; e < 256; e += 128) {
    int v = e >> 5;
    int l = e & 31;
    float gi = lds[0][v][l];
    float gf = lds[1][v][l];
    float gg = lds[2][v][l];
    float go = lds[3][v][l];

    int M   = v + ((l & 16) ? 8 : 0);
    int N   = l & 15;
    int row = m0 + M;           // batch index
    int col = j * 16 + N;       // hidden index

    float it = sigmoidf_fast(gi);
    float ft = sigmoidf_fast(gf);
    float gt = tanhf(gg);
    float ot = sigmoidf_fast(go);

    size_t ci = (size_t)row * LH + col;
    float cn = ft * c[ci] + it * gt;
    c[ci] = cn;
    float hn = ot * tanhf(cn);

    hseq[((size_t)row * LS + t) * LH + col] = hn;
    h_out[ci] = (__bf16)hn;
    if (h_final) h_final[ci] = hn;
  }
}

// ---------------------------------------------------------------- launch

extern "C" void kernel_launch(void* const* d_in, const int* in_sizes, int n_in,
                              void* d_out, int out_size, void* d_ws, size_t ws_size,
                              hipStream_t stream) {
  (void)in_sizes; (void)n_in; (void)out_size; (void)ws_size;

  const float* x    = (const float*)d_in[0];  // (B,S,D)
  const float* wih  = (const float*)d_in[1];  // (D,4H)
  const float* whh  = (const float*)d_in[2];  // (H,4H)
  const float* bias = (const float*)d_in[3];  // (4H)

  float* out     = (float*)d_out;
  float* hseq    = out;                              // (B,S,H)
  float* h_final = out + (size_t)LB * LS * LH;       // (B,H)
  float* c       = h_final + (size_t)LB * LH;        // (B,H), running cell state

  // workspace carve-up (bf16): x copy, packed weights, h ping-pong
  __bf16* w  = (__bf16*)d_ws;
  __bf16* xb     = w;  w += (size_t)LB * LS * LD;    // 16.8M elems
  __bf16* wih_pk = w;  w += (size_t)LD * LG;         //  2.1M elems
  __bf16* whh_pk = w;  w += (size_t)LH * LG;         //  4.2M elems
  __bf16* h0     = w;  w += (size_t)LB * LH;
  __bf16* h1     = w;  w += (size_t)LB * LH;

  const int TPB = 256;
  {
    size_t n = (size_t)LB * LS * LD;
    f32_to_bf16_kernel<<<(unsigned)((n + TPB - 1) / TPB), TPB, 0, stream>>>(x, xb, n);
  }
  {
    size_t n = (size_t)LD * LG;
    pack_weights_kernel<<<(unsigned)((n + TPB - 1) / TPB), TPB, 0, stream>>>(
        wih, wih_pk, LD / 32, n);
  }
  {
    size_t n = (size_t)LH * LG;
    pack_weights_kernel<<<(unsigned)((n + TPB - 1) / TPB), TPB, 0, stream>>>(
        whh, whh_pk, LH / 32, n);
  }
  {
    size_t n = (size_t)LB * LH;
    zero_bf16_kernel<<<(unsigned)((n + TPB - 1) / TPB), TPB, 0, stream>>>(h0, n);
    zero_f32_kernel <<<(unsigned)((n + TPB - 1) / TPB), TPB, 0, stream>>>(c, n);
  }

  for (int t = 0; t < LS; ++t) {
    const __bf16* hi = (t & 1) ? h1 : h0;
    __bf16*       ho = (t & 1) ? h0 : h1;
    lstm_step_kernel<<<256, 128, 0, stream>>>(
        xb, wih_pk, whh_pk, bias, hi, ho, c, hseq,
        (t == LS - 1) ? h_final : nullptr, t);
  }
}
